// Att_R_89429809037647
// MI455X (gfx1250) — matrix operand order repaired and checked by reading, not compiled, not run on hardware
//
#include <hip/hip_runtime.h>

// ---------------------------------------------------------------------------
// Fused low-rank attention + Laplacian assembly for MI455X (gfx1250, wave32).
// One workgroup per environment (E=128). 512 threads = 16 waves.
// 8 batch rows in flight per round (2 waves each), 8 rounds = 64 rows/env.
// All matmuls via v_wmma_f32_16x16x32_f16 (f16 in, f32 accumulate).
// x/V/o are stored K-contiguous (transposed) in LDS so every B-fragment
// load merges into ds_load_b128; tanh is a branch-free exp/rcp sequence.
// ---------------------------------------------------------------------------

typedef __attribute__((ext_vector_type(16))) _Float16 v16h;
typedef __attribute__((ext_vector_type(8)))  float    v8f;

#define E_NUM   128
#define NA      64
#define D_DIM   64
#define RANK    32
#define H_DIM   16
#define THREADS 512
#define GROUPS  8          // concurrent batch rows per workgroup
#define ROUNDS  8          // GROUPS*ROUNDS = 64 rows per env

// ---- LDS layout (bytes) ---------------------------------------------------
// per-row region (20KB):
//   [0,8K)   : xT   f16 [n=64][d=64]  (aliased later by sc f32 [32][32] + oT)
//   [8K,12K) : Qh   f16 [r=32][n=64]  (aliased later by attn f16 [32][32])
//   [12K,16K): Kh   f16 [s=32][n=64]
//   [16K,20K): VT   f16 [n=64][s=32]
#define REG_BYTES (20 * 1024)
#define W_OFF     (GROUPS * REG_BYTES)          // 160KB: weights
#define AQ_OFF    (W_OFF)
#define AK_OFF    (W_OFF + 4 * 1024)
#define AV_OFF    (W_OFF + 8 * 1024)
#define AO_OFF    (W_OFF + 12 * 1024)
#define RSQ_OFF   (W_OFF + 13 * 1024)
#define RH2_OFF   (W_OFF + 14 * 1024)           // 174KB: Rh^2 f16 [64][16][64]
#define SMEM_BYTES (RH2_OFF + NA * H_DIM * NA * 2)   // 302KB total

// A-fragment (16xK tile, 16-bit): lane m = lane&15;
// half t<8 -> K = kb+t, t>=8 -> K = kb+t+8, kb = k0 + 8*(lane>=16).
// Reads are contiguous along K within each 8-half group -> ds_load_b128.
__device__ __forceinline__ v16h ldA(const _Float16* p, int m0, int k0,
                                    int stride, int lane) {
  int m  = m0 + (lane & 15);
  int kb = k0 + ((lane >> 4) << 3);
  const _Float16* row = p + m * stride;
  v16h a;
#pragma unroll
  for (int t = 0; t < 16; ++t) {
    int k = kb + (t < 8 ? t : t + 8);
    a[t] = row[k];
  }
  return a;
}

// B-fragment (Kx16 tile, 16-bit): lane n = n0 + (lane&15);
// half t<8 -> K = kb+t, t>=8 -> K = kb+t+8, kb = k0 + 8*(lane>=16).
// Callers pass kstride==1 (K-contiguous LDS layouts) -> ds_load_b128 x2.
__device__ __forceinline__ v16h ldB(const _Float16* p, int k0, int n0,
                                    int kstride, int nstride, int lane) {
  int n  = n0 + (lane & 15);
  int kb = k0 + ((lane >> 4) << 3);
  const _Float16* q = p + n * nstride + kb * kstride;
  v16h b;
#pragma unroll
  for (int t = 0; t < 16; ++t) b[t] = q[(t < 8 ? t : t + 8) * kstride];
  return b;
}

__device__ __forceinline__ v8f wmma16(v16h a, v16h b, v8f c) {
  return __builtin_amdgcn_wmma_f32_16x16x32_f16(false, a, false, b,
                                                (short)0, c, false, false);
}

// Branch-free tanh: sign(x) * (1 - e^{-2|x|}) / (1 + e^{-2|x|}).
// v_exp_f32 + v_rcp_f32, no EXEC divergence (safe to interleave with WMMA).
__device__ __forceinline__ float fast_tanh(float x) {
  float ax = __builtin_fabsf(x);
  float t  = __expf(-2.0f * ax);
  float y  = (1.0f - t) * __builtin_amdgcn_rcpf(1.0f + t);
  return __builtin_copysignf(y, x);
}

__global__ __launch_bounds__(THREADS, 1)
void att_r_fused(const float* __restrict__ xg, const float* __restrict__ Lg,
                 const float* __restrict__ Aq, const float* __restrict__ Ak,
                 const float* __restrict__ Av, const float* __restrict__ Ao,
                 float* __restrict__ out) {
  extern __shared__ char smem[];
  const int tid  = threadIdx.x;
  const int lane = tid & 31;
  const int wid  = tid >> 5;
  const int wsub = wid & 1;   // which of the 2 waves working this batch row
  const int g    = wid >> 1;  // batch-row group 0..7
  const int gl   = tid & 63;  // lane within the 2-wave group
  const int e    = blockIdx.x;

  _Float16* Aqh = (_Float16*)(smem + AQ_OFF);
  _Float16* Akh = (_Float16*)(smem + AK_OFF);
  _Float16* Avh = (_Float16*)(smem + AV_OFF);
  _Float16* Aoh = (_Float16*)(smem + AO_OFF);
  float*    rsq = (float*)(smem + RSQ_OFF);
  _Float16* Rh2 = (_Float16*)(smem + RH2_OFF);

  // Stage weights into LDS (f16), row-major [r][d] (A-frags read K-contiguous)
  for (int idx = tid; idx < RANK * D_DIM; idx += THREADS) {
    Aqh[idx] = (_Float16)Aq[idx];
    Akh[idx] = (_Float16)Ak[idx];
    Avh[idx] = (_Float16)Av[idx];
  }
  for (int idx = tid; idx < H_DIM * RANK; idx += THREADS)
    Aoh[idx] = (_Float16)Ao[idx];

  char* rbase = smem + g * REG_BYTES;
  _Float16* xT = (_Float16*)rbase;                 // [n=64][d=64]
  _Float16* Qh = (_Float16*)(rbase + 8 * 1024);    // [r=32][n=64]
  _Float16* Kh = (_Float16*)(rbase + 12 * 1024);   // [s=32][n=64]
  _Float16* VT = (_Float16*)(rbase + 16 * 1024);   // [n=64][s=32]
  float*    sc = (float*)rbase;                    // [32][32], aliases xT
  _Float16* oT = (_Float16*)(rbase + 4 * 1024);    // [n=64][r=32], aliases xT
  _Float16* at = Qh;                               // attn [32][32], aliases Qh

  for (int round = 0; round < ROUNDS; ++round) {
    const int i_local = round * GROUPS + g;
    const size_t bg   = (size_t)e * NA + i_local;
    __syncthreads();  // previous round fully consumed before xT overwrite

    // ---- stage 1: load x_b -> xT (f16, transposed); numN -> rsq[g] --------
    {
      const float4* src = (const float4*)(xg + bg * (D_DIM * NA));
#pragma unroll
      for (int it = 0; it < 16; ++it) {
        int f4 = it * 64 + gl;
        float4 v = src[f4];
        int d = (f4 * 4) >> 6;
        int n = (f4 * 4) & 63;
        xT[(n + 0) * D_DIM + d] = (_Float16)v.x;
        xT[(n + 1) * D_DIM + d] = (_Float16)v.y;
        xT[(n + 2) * D_DIM + d] = (_Float16)v.z;
        xT[(n + 3) * D_DIM + d] = (_Float16)v.w;
      }
      if (round + 1 < ROUNDS)  // global_prefetch_b8 for next round's tile
        __builtin_prefetch(xg + (bg + GROUPS) * (D_DIM * NA) + gl * 64, 0, 0);
      if (wsub == 0) {
        const float* Lr = Lg + bg * NA;
        float c = (Lr[lane] >= 1.0f ? 1.0f : 0.0f)
                + (Lr[lane + 32] >= 1.0f ? 1.0f : 0.0f);
#pragma unroll
        for (int d = 16; d >= 1; d >>= 1) c += __shfl_xor(c, d, 32);
        if (lane == 0) rsq[g] = rsqrtf(c + 1.0f);
      }
    }
    __syncthreads();

    // ---- stage 2: Q,K,V = tanh(W @ x); 24 tiles split 12/12 ---------------
    // Q,K stored row-major [r][n]; V stored transposed [n][s].
    for (int t = wsub * 12; t < wsub * 12 + 12; ++t) {
      int wsel  = t >> 3;
      int rtile = (t >> 2) & 1;
      int ntile = t & 3;
      const _Float16* W = (wsel == 0) ? Aqh : (wsel == 1) ? Akh : Avh;
      v8f acc = {};
#pragma unroll
      for (int dc = 0; dc < 2; ++dc) {
        v16h a = ldA(W, rtile * 16, dc * 32, D_DIM, lane);
        v16h b = ldB(xT, dc * 32, ntile * 16, 1, D_DIM, lane);  // K-contig
        acc = wmma16(a, b, acc);
      }
      int n  = ntile * 16 + (lane & 15);
      int mb = rtile * 16 + ((lane >> 4) << 3);
      if (wsel == 2) {
#pragma unroll
        for (int v = 0; v < 8; ++v)
          VT[n * RANK + (mb + v)] = (_Float16)fast_tanh(acc[v]);
      } else {
        _Float16* dst = (wsel == 0) ? Qh : Kh;
#pragma unroll
        for (int v = 0; v < 8; ++v)
          dst[(mb + v) * NA + n] = (_Float16)fast_tanh(acc[v]);
      }
    }
    __syncthreads();

    // ---- stage 3: scores = (Q K^T) * rsqrt(numN); wave owns row-tile wsub -
    {
      float rs = rsq[g];
#pragma unroll
      for (int st = 0; st < 2; ++st) {
        v8f acc = {};
#pragma unroll
        for (int nc = 0; nc < 2; ++nc) {
          v16h a = ldA(Qh, wsub * 16, nc * 32, NA, lane);
          v16h b = ldB(Kh, nc * 32, st * 16, 1, NA, lane);  // K^T, K-contig
          acc = wmma16(a, b, acc);
        }
        int s  = st * 16 + (lane & 15);
        int mb = wsub * 16 + ((lane >> 4) << 3);
#pragma unroll
        for (int v = 0; v < 8; ++v) sc[(mb + v) * RANK + s] = acc[v] * rs;
      }
    }
    __syncthreads();

    // ---- stage 4: softmax rows -> attn f16 (overwrites Qh region) ---------
    {
      int r = wsub * 16 + (lane & 15);
      const float* srow = sc + r * RANK + ((lane >> 4) << 4);
      float vals[16];
      float mx = -3.4e38f;
#pragma unroll
      for (int t = 0; t < 16; ++t) { vals[t] = srow[t]; mx = fmaxf(mx, vals[t]); }
      mx = fmaxf(mx, __shfl_xor(mx, 16, 32));
      float sum = 0.0f;
#pragma unroll
      for (int t = 0; t < 16; ++t) { vals[t] = __expf(vals[t] - mx); sum += vals[t]; }
      sum += __shfl_xor(sum, 16, 32);
      float inv = __builtin_amdgcn_rcpf(sum);
      _Float16* arow = at + r * RANK + ((lane >> 4) << 4);
#pragma unroll
      for (int t = 0; t < 16; ++t) arow[t] = (_Float16)(vals[t] * inv);
    }
    __syncthreads();

    // ---- stage 5: o = attn @ V -> oT (f16, transposed [n][r]) -------------
    {
      v16h a = ldA(at, wsub * 16, 0, RANK, lane);
#pragma unroll
      for (int nt = 0; nt < 4; ++nt) {
        v8f acc = {};
        v16h b = ldB(VT, 0, nt * 16, 1, RANK, lane);  // B[k=s][n], K-contig
        acc = wmma16(a, b, acc);
        int n  = nt * 16 + (lane & 15);
        int mb = wsub * 16 + ((lane >> 4) << 3);
#pragma unroll
        for (int v = 0; v < 8; ++v) oT[n * RANK + (mb + v)] = (_Float16)acc[v];
      }
    }
    __syncthreads();

    // ---- stage 6: Rh2 = tanh(Ao @ o)^2 -> LDS [i][h][n] -------------------
    {
      v16h a = ldA(Aoh, 0, 0, RANK, lane);
#pragma unroll
      for (int u = 0; u < 2; ++u) {
        int nt = wsub * 2 + u;
        v8f acc = {};
        v16h b = ldB(oT, 0, nt * 16, 1, RANK, lane);  // B[k=r][n], K-contig
        acc = wmma16(a, b, acc);
        int n  = nt * 16 + (lane & 15);
        int hb = (lane >> 4) << 3;
#pragma unroll
        for (int v = 0; v < 8; ++v) {
          float th = fast_tanh(acc[v]);
          Rh2[i_local * (H_DIM * NA) + (hb + v) * NA + n] = (_Float16)(th * th);
        }
      }
    }
  }
  __syncthreads();

  // ---- Phase B: permutation + symmetrize + Laplacian, write [256][256] ----
  // Rupper[r][q] = Rh2[i, j>>2, (j&3)*16+k] + (r<->q term), with
  // r = hr*128+2i+pr, q = hc*128+2j+pc, k = 8hr+4hc+2pr+pc.
  // Rout off-diag = -Rupper; Rout[r][r] = full row sum of Rupper.
  float* outE = out + (size_t)e * (4 * NA) * (4 * NA);
  for (int rr = 0; rr < 16; ++rr) {
    int r  = wid * 16 + rr;
    int i  = (r & 127) >> 1, pr = r & 1, hr = r >> 7;
    float* orow = outE + r * 256;
    float sum = 0.0f;
#pragma unroll
    for (int m = 0; m < 8; ++m) {
      int q  = lane + (m << 5);
      int j  = (q & 127) >> 1, pc = q & 1, hc = q >> 7;
      int k1 = hr * 8 + hc * 4 + pr * 2 + pc;
      int k2 = hc * 8 + hr * 4 + pc * 2 + pr;
      float A1 = (float)Rh2[i * 1024 + ((j >> 2) << 6) + ((j & 3) << 4) + k1];
      float A2 = (float)Rh2[j * 1024 + ((i >> 2) << 6) + ((i & 3) << 4) + k2];
      float val = A1 + A2;
      sum += val;
      if (q != r) orow[q] = -val;
    }
#pragma unroll
    for (int d = 16; d >= 1; d >>= 1) sum += __shfl_xor(sum, d, 32);
    if (lane == (r & 31)) orow[r] = sum;
  }
}

extern "C" void kernel_launch(void* const* d_in, const int* in_sizes, int n_in,
                              void* d_out, int out_size, void* d_ws, size_t ws_size,
                              hipStream_t stream) {
  (void)in_sizes; (void)n_in; (void)d_ws; (void)ws_size; (void)out_size;
  const float* x  = (const float*)d_in[0];
  const float* L  = (const float*)d_in[1];
  const float* Aq = (const float*)d_in[2];
  const float* Ak = (const float*)d_in[3];
  const float* Av = (const float*)d_in[4];
  const float* Ao = (const float*)d_in[5];
  float* out = (float*)d_out;
  att_r_fused<<<dim3(E_NUM), dim3(THREADS), SMEM_BYTES, stream>>>(
      x, L, Aq, Ak, Av, Ao, out);
}
